// HiResPrecipNet_CNN_GNN_new_89223650607421
// MI455X (gfx1250) — compile-verified, matrix operand order checked by reading, not run on hardware
//
#include <hip/hip_runtime.h>
#include <hip/hip_bf16.h>
#include <math.h>

#define NLOW   10000
#define NHIGH  100000
#define EL2H   900000
#define EHH    600000
#define BN_EPS 1e-5f

typedef __attribute__((ext_vector_type(16))) __bf16       v16bf;
typedef __attribute__((ext_vector_type(8)))  __bf16       v8bf;
typedef __attribute__((ext_vector_type(8)))  float        v8f;
typedef __attribute__((ext_vector_type(4)))  unsigned int v4u;
typedef __attribute__((ext_vector_type(8)))  int          v8i;
typedef __attribute__((ext_vector_type(4)))  int          v4i;

static inline long cdivl(long a, long b) { return (a + b - 1) / b; }

// ---------------------------------------------------------------------------
// helpers
// ---------------------------------------------------------------------------
__device__ __forceinline__ void atomic_max_f32(float* addr, float val) {
  if (val >= 0.0f) atomicMax((int*)addr, __float_as_int(val));
  else             atomicMin((unsigned int*)addr, (unsigned int)__float_as_int(val));
}

__global__ void fill_f32(float* __restrict__ p, float v, long n) {
  long i = (long)blockIdx.x * blockDim.x + threadIdx.x;
  if (i < n) p[i] = v;
}

// ---------------------------------------------------------------------------
// TDM: async 2D f32 tile load Global -> LDS (D# per CDNA5 ISA ch.8).
// Tile-relative extents give correct OOB zero-fill; pad fields add an LDS
// pitch stagger (pad_interval code 4 = every 32 dwords, pad_amount code 3 =
// 4 dwords) so each 32-f32 tile row lands at pitch 36 f32 (144B, 16B-aligned).
// ---------------------------------------------------------------------------
__device__ __forceinline__ void tdm_load_2d_f32(unsigned lds_off, const float* gaddr,
                                                unsigned tensor_d0, unsigned tensor_d1,
                                                unsigned tile_d0, unsigned tile_d1,
                                                unsigned long long stride0) {
  unsigned long long ga = (unsigned long long)(const void*)gaddr;
  v4u g0;
  g0[0] = 1u;                                              // count=1, user D#
  g0[1] = lds_off;                                         // lds_addr (bytes)
  g0[2] = (unsigned)(ga & 0xFFFFFFFFu);                    // global_addr[31:0]
  g0[3] = (unsigned)((ga >> 32) & 0x1FFFFFFu) | (2u << 30); // addr[56:32] | type=2
  v8i g1;
  g1[0] = (int)((2u << 16) | (1u << 20) | (4u << 22) | (3u << 25)); // 4B elems, pad on
  g1[1] = (int)((tensor_d0 & 0xFFFFu) << 16);              // tensor_dim0 lo
  g1[2] = (int)(((tensor_d0 >> 16) & 0xFFFFu) | ((tensor_d1 & 0xFFFFu) << 16));
  g1[3] = (int)(((tensor_d1 >> 16) & 0xFFFFu) | (tile_d0 << 16));
  g1[4] = (int)(tile_d1 & 0xFFFFu);                        // tile_dim1, tile_dim2=0
  g1[5] = (int)(unsigned)(stride0 & 0xFFFFFFFFu);          // tensor_dim0_stride lo
  g1[6] = (int)(unsigned)((stride0 >> 32) & 0xFFFFu);
  g1[7] = 0;
  v4i gz = {0, 0, 0, 0};
#if __clang_major__ >= 23
  v8i gz8 = {0, 0, 0, 0, 0, 0, 0, 0};
  __builtin_amdgcn_tensor_load_to_lds(g0, g1, gz, gz, gz8, 0);
#else
  __builtin_amdgcn_tensor_load_to_lds(g0, g1, gz, gz, 0);
#endif
}

// ---------------------------------------------------------------------------
// CNN: direct conv3d (kernel depth 1, H kernel 5 unpadded, W kernel 3 pad padW)
// ---------------------------------------------------------------------------
__global__ void conv3d_kernel(const float* __restrict__ in, const float* __restrict__ w,
                              const float* __restrict__ b, float* __restrict__ out,
                              int N, int Ci, int Co, int Hin, int Hout, int Win, int Wout,
                              int padW) {
  long idx = (long)blockIdx.x * blockDim.x + threadIdx.x;
  long total = (long)N * Co * 5 * Hout * Wout;
  if (idx >= total) return;
  int wo = (int)(idx % Wout); long t = idx / Wout;
  int ho = (int)(t % Hout);   t /= Hout;
  int d  = (int)(t % 5);      t /= 5;
  int co = (int)(t % Co);
  int n  = (int)(t / Co);
  float acc = b[co];
  for (int ci = 0; ci < Ci; ++ci) {
    const float* ip = in + (((long)n * Ci + ci) * 5 + d) * (long)(Hin * Win);
    const float* wp = w + ((long)co * Ci + ci) * 15;
#pragma unroll
    for (int kh = 0; kh < 5; ++kh) {
      int hi = ho + kh;
#pragma unroll
      for (int kw = 0; kw < 3; ++kw) {
        int wi = wo + kw - padW;
        if (wi >= 0 && wi < Win) acc += ip[hi * Win + wi] * wp[kh * 3 + kw];
      }
    }
  }
  out[idx] = acc;
}

// ---------------------------------------------------------------------------
// BatchNorm over (N, S) per channel C; layout ((n*C + c)*S + s). S=1 for 2D.
// ---------------------------------------------------------------------------
__global__ void bn_stats_kernel(const float* __restrict__ x, float* __restrict__ stats,
                                int N, int C, int S) {
  __shared__ float ls[256], lq[256];
  int c = blockIdx.x;
  long total = (long)N * S;
  float s = 0.f, q = 0.f;
  for (long i = (long)blockIdx.y * blockDim.x + threadIdx.x; i < total;
       i += (long)gridDim.y * blockDim.x) {
    long n = i / S, sp = i % S;
    float v = x[((n * C) + c) * S + sp];
    s += v; q += v * v;
  }
  ls[threadIdx.x] = s; lq[threadIdx.x] = q;
  __syncthreads();
  for (int o = 128; o > 0; o >>= 1) {
    if ((int)threadIdx.x < o) {
      ls[threadIdx.x] += ls[threadIdx.x + o];
      lq[threadIdx.x] += lq[threadIdx.x + o];
    }
    __syncthreads();
  }
  if (threadIdx.x == 0) {
    atomicAdd(&stats[2 * c], ls[0]);
    atomicAdd(&stats[2 * c + 1], lq[0]);
  }
}

__global__ void bn_apply_kernel(float* __restrict__ x, const float* __restrict__ g,
                                const float* __restrict__ bt, const float* __restrict__ stats,
                                int N, int C, int S, int relu) {
  long idx = (long)blockIdx.x * blockDim.x + threadIdx.x;
  long total = (long)N * C * S;
  if (idx >= total) return;
  int c = (int)((idx / S) % C);
  float cntf = (float)N * (float)S;
  float m = stats[2 * c] / cntf;
  float v = stats[2 * c + 1] / cntf - m * m;
  float y = (x[idx] - m) * rsqrtf(v + BN_EPS) * g[c] + bt[c];
  if (relu) y = fmaxf(y, 0.f);
  x[idx] = y;
}

// ---------------------------------------------------------------------------
// max pool (1,1,1,5,3) stride (1,1,1,2,1) over (N,32,5,5,3) -> (N,160) flat
// ---------------------------------------------------------------------------
__global__ void maxpool_kernel(const float* __restrict__ a, float* __restrict__ pooled, int N) {
  long idx = (long)blockIdx.x * blockDim.x + threadIdx.x;
  if (idx >= (long)N * 160) return;
  int d = (int)(idx % 5);
  int c = (int)((idx / 5) % 32);
  int n = (int)(idx / 160);
  const float* p = a + (((long)n * 32 + c) * 5 + d) * 15;
  float m = p[0];
#pragma unroll
  for (int i = 1; i < 15; ++i) m = fmaxf(m, p[i]);
  pooled[(long)n * 160 + c * 5 + d] = m;
}

// ---------------------------------------------------------------------------
// WMMA bf16 GEMM: C(MxN) = relu?( A(MxK) @ B(KxN) + bias(N) )
// 128 threads = 4 waves; block tile 64x16; K unrolled via NSTEP template.
// A slab: TDM async double-buffered (f32, pitch 36), wave0 issues + waits
// tensorcnt. B slab: per-thread one k-row / 4 consecutive cols, phase-split
// load-then-store so all global loads batch under one wait.
// ---------------------------------------------------------------------------
template <int NSTEP>
__global__ void wmma_gemm_kernel(const float* __restrict__ A, const float* __restrict__ B,
                                 const float* __restrict__ bias, float* __restrict__ C,
                                 int M, int N, int K, int relu) {
  constexpr int APF = 36;                       // f32 pitch from TDM pad fields
  constexpr int BP  = 40;                       // bf16 pitch (80B)
  __shared__ __align__(16) float  Af[2][64 * APF];
  __shared__ __align__(16) __bf16 Bs[16 * BP];
  const int t    = threadIdx.x;
  const int lane = t & 31;
  const int wv   = t >> 5;
  const int hl   = lane & 15;
  const int hi   = lane >> 4;
  const int bm   = blockIdx.x * 64;
  const int n0   = blockIdx.y * 16;

  const unsigned rowsLeft = (unsigned)max(M - bm, 0);
  const float* atile = A + (long)bm * K;

  if (wv == 0)
    tdm_load_2d_f32((unsigned)(size_t)&Af[0][0], atile, (unsigned)K, rowsLeft, 32, 64,
                    (unsigned long long)K);

  // B staging geometry: thread t covers k-row kk, 4 consecutive cols col0..col0+3
  const int kk   = t >> 2;        // 0..31
  const int col0 = (t * 4) & 15;  // 0,4,8,12

  v8f acc = {0.f, 0.f, 0.f, 0.f, 0.f, 0.f, 0.f, 0.f};

#pragma unroll
  for (int stp = 0; stp < NSTEP; ++stp) {
    const int k0 = stp * 32;
    const int kg = k0 + kk;
    const bool krowOK = (kg < K);
    const float* brow = B + (long)min(kg, K - 1) * N;
    if (stp + 1 < NSTEP)
      __builtin_prefetch(brow + 32 * N - ((long)min(kg, K - 1) - min(kg + 32, K - 1)) * 0 +
                             min(n0 + col0, N - 1), 0, 1);
    // phase 1: batch the 4 consecutive-column loads (single wait)
    float bv[4];
#pragma unroll
    for (int j = 0; j < 4; ++j) bv[j] = brow[min(n0 + col0 + j, N - 1)];
    // phase 2: convert + transposed LDS stores
#pragma unroll
    for (int j = 0; j < 4; ++j)
      Bs[(col0 + j) * BP + kk] =
          (__bf16)((krowOK && (n0 + col0 + j) < N) ? bv[j] : 0.f);

    // ---- pipeline next A slab via TDM, wait for current ----
    if (wv == 0) {
      if (stp + 1 < NSTEP) {
        tdm_load_2d_f32((unsigned)(size_t)&Af[(stp + 1) & 1][0], atile + (k0 + 32),
                        (unsigned)(K - (k0 + 32)), rowsLeft, 32, 64,
                        (unsigned long long)K);
        __builtin_amdgcn_s_wait_tensorcnt(1);   // current slab done, next in flight
      } else {
        __builtin_amdgcn_s_wait_tensorcnt(0);
      }
    }
    __syncthreads();

    // ---- fragments: ISA 7.12.2 layout; lane row=hl, K-group select=hi ----
    const int row = wv * 16 + hl;
    const float* ar = &Af[stp & 1][row * APF];
    v16bf a;
#pragma unroll
    for (int j = 0; j < 8; ++j) a[j] = (__bf16)ar[hi * 8 + j];          // k=kb..kb+7
#pragma unroll
    for (int j = 0; j < 8; ++j) a[8 + j] = (__bf16)ar[hi * 8 + 16 + j]; // kb+16..kb+23
    v8bf b0 = *(const v8bf*)&Bs[hl * BP + hi * 8];
    v8bf b1 = *(const v8bf*)&Bs[hl * BP + 16 + hi * 8];
    v16bf b = __builtin_shufflevector(b0, b1, 0, 1, 2, 3, 4, 5, 6, 7, 8, 9, 10, 11, 12,
                                      13, 14, 15);
    acc = __builtin_amdgcn_wmma_f32_16x16x32_bf16(false, a, false, b, (short)0, acc,
                                                  false, false);
    __syncthreads();
  }

  // ---- epilogue: single bias load, running pointer, interior fast path ----
  const int m0 = bm + wv * 16;
  const int col = n0 + hl;
  const bool colOK = (col < N);
  const float bcol = bias[min(col, N - 1)];
  float* cp = C + (long)(m0 + hi * 8) * N + col;
  if (m0 + 15 < M && colOK) {
#pragma unroll
    for (int v = 0; v < 8; ++v) {
      float r = acc[v] + bcol;
      if (relu) r = fmaxf(r, 0.f);
      *cp = r;
      cp += N;
    }
  } else {
#pragma unroll
    for (int v = 0; v < 8; ++v) {
      int m = m0 + hi * 8 + v;
      if (m < M && colOK) {
        float r = acc[v] + bcol;
        if (relu) r = fmaxf(r, 0.f);
        *cp = r;
      }
      cp += N;
    }
  }
}

// K==1 projection: plain outer product (WMMA would waste a 32-deep MAC)
__global__ void outer_k1_kernel(const float* __restrict__ x, const float* __restrict__ w,
                                const float* __restrict__ bias, float* __restrict__ C,
                                int M, int N) {
  long i = (long)blockIdx.x * blockDim.x + threadIdx.x;
  if (i >= (long)M * N) return;
  int n = (int)(i % N);
  long m = i / N;
  C[i] = x[m] * w[n] + bias[n];
}

// ---------------------------------------------------------------------------
// GATv2 edge passes. Edges e < E come from (src,dst); e >= E are self-loops
// (node id = e - E). Wave per (edge,head): lanes stride features (coalesced).
// ---------------------------------------------------------------------------
__global__ void gat_pass1(const float* __restrict__ xl, const float* __restrict__ xr,
                          const float* __restrict__ att, const int* __restrict__ src,
                          const int* __restrict__ dst, int E, int Eext, int H, int O,
                          float* __restrict__ ebuf, float* __restrict__ mbuf) {
  long wid = ((long)blockIdx.x * blockDim.x + threadIdx.x) >> 5;
  int lane = threadIdx.x & 31;
  if (wid >= (long)Eext * H) return;
  int h = (int)(wid % H);
  long e = wid / H;
  int s, d;
  if (e < E) { s = src[e]; d = dst[e]; } else { s = d = (int)(e - E); }
  const float* pl = xl + ((long)s * H + h) * O;
  const float* pr = xr + ((long)d * H + h) * O;
  const float* pa = att + h * O;
  float acc = 0.f;
  for (int o = lane; o < O; o += 32) {
    float t = pl[o] + pr[o];
    t = t > 0.f ? t : 0.2f * t;   // leaky_relu(0.2)
    acc += t * pa[o];
  }
#pragma unroll
  for (int m = 16; m >= 1; m >>= 1) acc += __shfl_xor(acc, m, 32);
  if (lane == 0) {
    ebuf[wid] = acc;
    atomic_max_f32(&mbuf[(long)d * H + h], acc);
  }
}

__global__ void gat_pass2(const int* __restrict__ dst, int E, int Eext, int H,
                          float* __restrict__ ebuf, const float* __restrict__ mbuf,
                          float* __restrict__ den, float* __restrict__ cnt) {
  long tid = (long)blockIdx.x * blockDim.x + threadIdx.x;
  if (tid >= (long)Eext * H) return;
  int h = (int)(tid % H);
  long e = tid / H;
  int d = (e < E) ? dst[e] : (int)(e - E);
  float a = __expf(ebuf[tid] - mbuf[(long)d * H + h]);
  ebuf[tid] = a;
  atomicAdd(&den[(long)d * H + h], a);
  if (h == 0) atomicAdd(&cnt[d], 1.0f);
}

__global__ void gat_pass3(const float* __restrict__ xl, const int* __restrict__ src,
                          const int* __restrict__ dst, int E, int Eext, int H, int O,
                          const float* __restrict__ ebuf, const float* __restrict__ den,
                          float* __restrict__ sbuf) {
  long wid = ((long)blockIdx.x * blockDim.x + threadIdx.x) >> 5;
  int lane = threadIdx.x & 31;
  if (wid >= (long)Eext * H) return;
  int h = (int)(wid % H);
  long e = wid / H;
  int s, d;
  if (e < E) { s = src[e]; d = dst[e]; } else { s = d = (int)(e - E); }
  float alpha = ebuf[wid] / fmaxf(den[(long)d * H + h], 1e-16f);
  const float* pl = xl + ((long)s * H + h) * O;
  float* ps = sbuf + ((long)d * H + h) * O;
  for (int o = lane; o < O; o += 32) atomicAdd(&ps[o], pl[o] * alpha);
}

__global__ void gat_finalize(const float* __restrict__ sbuf, const float* __restrict__ cnt,
                             const float* __restrict__ bias, float* __restrict__ out,
                             int Nn, int HO, int ld, int c0, int relu) {
  long tid = (long)blockIdx.x * blockDim.x + threadIdx.x;
  if (tid >= (long)Nn * HO) return;
  int n = (int)(tid / HO);
  int c = (int)(tid % HO);
  float r = sbuf[tid] / fmaxf(cnt[n], 1.f) + bias[c];
  if (relu) r = fmaxf(r, 0.f);
  out[(long)n * ld + c0 + c] = r;
}

__global__ void set_col0(float* __restrict__ h, const float* __restrict__ z, int Nn, int ld) {
  long i = (long)blockIdx.x * blockDim.x + threadIdx.x;
  if (i < Nn) h[i * ld] = z[i];
}

// ---------------------------------------------------------------------------
// host orchestration
// ---------------------------------------------------------------------------
extern "C" void kernel_launch(void* const* d_in, const int* in_sizes, int n_in,
                              void* d_out, int out_size, void* d_ws, size_t ws_size,
                              hipStream_t stream) {
  (void)in_sizes; (void)n_in; (void)out_size; (void)ws_size;
  const float* x_low   = (const float*)d_in[0];
  const float* x_high  = (const float*)d_in[1];
  const float* z_std   = (const float*)d_in[2];
  const int*   src_l2h = (const int*)d_in[3];
  const int*   dst_l2h = (const int*)d_in[4];
  const int*   edge_hh = (const int*)d_in[5];
  // params flattened by sorted key (jax pytree order), starting at d_in[6]:
  // 6..15 bn{0..4}_{b,g} | 16..20 cW | 21..25 cb | 26..30 cbeta | 31..35 cg
  // 36 dW 37 db | 38..43 down{Wl,Wr,att,bias,bl,br} | 44..67 g0..g3 | 68..73 g4
  // 74..76 pW0..2 | 77..79 pb0..2
  auto P = [&](int i) { return (const float*)d_in[i]; };

  float* ws = (float*)d_ws;
  const size_t PER = 14400;                        // max per-sample activation
  float* cnnA   = ws;
  float* cnnB   = ws + (size_t)NLOW * PER;
  float* pooled = cnnB + (size_t)NLOW * PER;       // (10000,160)
  float* enc    = pooled + (size_t)NLOW * 160;     // (10000,32)
  float* stats  = enc + (size_t)NLOW * 32;         // 2*128 floats
  // GNN region aliases cnnA (CNN activations dead once pooled)
  float* hb0 = ws;
  float* hb1 = hb0 + (size_t)NHIGH * 128;
  float* xlp = hb1 + (size_t)NHIGH * 128;
  float* xrp = xlp + (size_t)NHIGH * 128;
  float* sb  = xrp + (size_t)NHIGH * 128;
  float* eb  = sb + (size_t)NHIGH * 128;           // 1.5M floats (max Eext*H)
  float* mm  = eb + 1500000;
  float* dd  = mm + (size_t)NHIGH * 2;
  float* cn  = dd + (size_t)NHIGH * 2;

  auto gsz = [](long n) { return (unsigned)cdivl(n, 256); };
  auto launch_fill = [&](float* p, float v, long n) {
    fill_f32<<<gsz(n), 256, 0, stream>>>(p, v, n);
  };
  auto launch_gemm = [&](const float* A, const float* B, const float* bias, float* Cm,
                         int M, int Nn, int K, int relu) {
    if (K == 1) {
      outer_k1_kernel<<<gsz((long)M * Nn), 256, 0, stream>>>(A, B, bias, Cm, M, Nn);
      return;
    }
    dim3 g((unsigned)cdivl(M, 64), (unsigned)cdivl(Nn, 16));
    switch ((int)cdivl(K, 32)) {
      case 1: wmma_gemm_kernel<1><<<g, 128, 0, stream>>>(A, B, bias, Cm, M, Nn, K, relu); break;
      case 2: wmma_gemm_kernel<2><<<g, 128, 0, stream>>>(A, B, bias, Cm, M, Nn, K, relu); break;
      case 3: wmma_gemm_kernel<3><<<g, 128, 0, stream>>>(A, B, bias, Cm, M, Nn, K, relu); break;
      case 4: wmma_gemm_kernel<4><<<g, 128, 0, stream>>>(A, B, bias, Cm, M, Nn, K, relu); break;
      default: wmma_gemm_kernel<5><<<g, 128, 0, stream>>>(A, B, bias, Cm, M, Nn, K, relu); break;
    }
  };
  auto launch_bn = [&](float* x, const float* g, const float* bt, int N, int C, int S,
                       int relu) {
    launch_fill(stats, 0.f, 2L * C);
    bn_stats_kernel<<<dim3((unsigned)C, 64), 256, 0, stream>>>(x, stats, N, C, S);
    long tot = (long)N * C * S;
    bn_apply_kernel<<<gsz(tot), 256, 0, stream>>>(x, g, bt, stats, N, C, S, relu);
  };
  auto run_gat = [&](const float* xsrc, int nsrc, int dinS, const float* xdst, int ndst,
                     int dinD, const int* src, const int* dst, int E, int Eext, int H,
                     int O, int pbase, float* outp, int ld, int c0, int relu) {
    const float* Wl = P(pbase + 0); const float* Wr = P(pbase + 1);
    const float* att = P(pbase + 2); const float* bias = P(pbase + 3);
    const float* bl = P(pbase + 4); const float* br = P(pbase + 5);
    int HO = H * O;
    launch_gemm(xsrc, Wl, bl, xlp, nsrc, HO, dinS, 0);
    launch_gemm(xdst, Wr, br, xrp, ndst, HO, dinD, 0);
    launch_fill(mm, -INFINITY, (long)ndst * H);
    launch_fill(dd, 0.f, (long)ndst * H);
    launch_fill(sb, 0.f, (long)ndst * HO);
    launch_fill(cn, 0.f, ndst);
    long EH = (long)Eext * H;
    gat_pass1<<<gsz(EH * 32), 256, 0, stream>>>(xlp, xrp, att, src, dst, E, Eext, H, O, eb, mm);
    gat_pass2<<<gsz(EH), 256, 0, stream>>>(dst, E, Eext, H, eb, mm, dd, cn);
    gat_pass3<<<gsz(EH * 32), 256, 0, stream>>>(xlp, src, dst, E, Eext, H, O, eb, dd, sb);
    gat_finalize<<<gsz((long)ndst * HO), 256, 0, stream>>>(sb, cn, bias, outp, ndst, HO,
                                                           ld, c0, relu);
  };

  // ---- CNN ----
  struct CC { int Ci, Co, Hin, Hout, Win, Wout, pad; };
  const CC cfg[5] = {{1, 8, 25, 21, 5, 5, 1},  {8, 16, 21, 17, 5, 5, 1},
                     {16, 32, 17, 13, 5, 5, 1}, {32, 64, 13, 9, 5, 5, 1},
                     {64, 32, 9, 5, 5, 3, 0}};
  const float* cur_in = x_low;
  float* bufs[2] = {cnnA, cnnB};
  for (int i = 0; i < 5; ++i) {
    float* o = bufs[i & 1];
    long tot = (long)NLOW * cfg[i].Co * 5 * cfg[i].Hout * cfg[i].Wout;
    conv3d_kernel<<<gsz(tot), 256, 0, stream>>>(cur_in, P(16 + i), P(21 + i), o, NLOW,
                                                cfg[i].Ci, cfg[i].Co, cfg[i].Hin,
                                                cfg[i].Hout, cfg[i].Win, cfg[i].Wout,
                                                cfg[i].pad);
    launch_bn(o, P(31 + i), P(26 + i), NLOW, cfg[i].Co, 5 * cfg[i].Hout * cfg[i].Wout, 1);
    cur_in = o;
  }
  maxpool_kernel<<<gsz((long)NLOW * 160), 256, 0, stream>>>(cnnA, pooled, NLOW);
  launch_gemm(pooled, P(36), P(37), enc, NLOW, 32, 160, 1);   // dense + relu

  // ---- down GAT (low -> high), output into hcat cols 1..64; col0 = z_std ----
  run_gat(enc, NLOW, 32, x_high, NHIGH, 1, src_l2h, dst_l2h, EL2H, EL2H, 1, 64, 38,
          hb0, 65, 1, 0);
  set_col0<<<gsz(NHIGH), 256, 0, stream>>>(hb0, z_std, NHIGH, 65);
  launch_bn(hb0, P(7), P(6), NHIGH, 65, 1, 0);                // bn0, no relu

  // ---- high-high GAT stack (self loops synthesized in-kernel) ----
  const int* ehsrc = edge_hh;
  const int* ehdst = edge_hh + EHH;
  float* cur = hb0;
  float* nxt = hb1;
  int din = 65;
  for (int i = 0; i < 4; ++i) {
    run_gat(cur, NHIGH, din, cur, NHIGH, din, ehsrc, ehdst, EHH, EHH + NHIGH, 2, 64,
            44 + 6 * i, nxt, 128, 0, 0);
    launch_bn(nxt, P(9 + 2 * i), P(8 + 2 * i), NHIGH, 128, 1, 1);  // bn{i+1} + relu
    float* t = cur; cur = nxt; nxt = t;
    din = 128;
  }
  run_gat(cur, NHIGH, 128, cur, NHIGH, 128, ehsrc, ehdst, EHH, EHH + NHIGH, 1, 64, 68,
          nxt, 64, 0, 1);                                     // g4 + relu

  // ---- MLP head ----
  launch_gemm(nxt, P(74), P(77), xlp, NHIGH, 64, 64, 1);
  launch_gemm(xlp, P(75), P(78), xrp, NHIGH, 32, 64, 1);
  launch_gemm(xrp, P(76), P(79), (float*)d_out, NHIGH, 1, 32, 0);
}
